// HierarchicalGCN_52123723104292
// MI455X (gfx1250) — compile-verified, matrix operand order checked by reading, not run on hardware
//
#include <hip/hip_runtime.h>
#include <hip/hip_bf16.h>
#include <math.h>

// ---------------------------------------------------------------------------
// CDNA5 (gfx1250) WMMA types
// ---------------------------------------------------------------------------
typedef __attribute__((ext_vector_type(16))) __bf16 v16bf;
typedef __attribute__((ext_vector_type(8)))  __bf16 v8bf;
typedef __attribute__((ext_vector_type(8)))  float  v8f;

static inline int  cdiv (int a, int b)   { return (a + b - 1) / b; }
static inline long cdivl(long a, long b) { return (a + b - 1) / b; }

#define MAX_KPAD 320          // max padded K in this model (IN_DIM 300 -> 320)
#define LDS_PAD  8            // +8 bf16 per row: breaks 640B stride bank aliasing

// ---------------------------------------------------------------------------
// Pre-conversion kernels: make GEMM operands branch-free.
//   Ab[Mpad128,Kpad] = bf16(A[M,K])   zero padded (rows to x128, K to x32)
//   Wt[Npad64,Kpad]  = bf16(W[K,N])^T zero padded (column-major weights)
// ---------------------------------------------------------------------------
__global__ void cvt_a_bf16_k(const float* __restrict__ A, __bf16* __restrict__ Ab,
                             int M, int K, int Kpad, long total) {
  long i = (long)blockIdx.x * blockDim.x + threadIdx.x;
  if (i >= total) return;
  const int m = (int)(i / Kpad);
  const int k = (int)(i % Kpad);
  const float v = (m < M && k < K) ? A[(long)m * K + k] : 0.0f;
  Ab[i] = (__bf16)v;
}
__global__ void cvt_wt_bf16_k(const float* __restrict__ W, __bf16* __restrict__ Wt,
                              int K, int N, int Kpad, long total) {
  long i = (long)blockIdx.x * blockDim.x + threadIdx.x;
  if (i >= total) return;
  const int n = (int)(i / Kpad);
  const int k = (int)(i % Kpad);
  const float v = (n < N && k < K) ? W[(long)k * N + n] : 0.0f;
  Wt[i] = (__bf16)v;
}

// ---------------------------------------------------------------------------
// bf16 WMMA GEMM, block tile 128(M) x 64(N), 8 waves/block:
//  - B panel [64 x Kpad] staged ONCE per block into LDS (<=40KB of 320KB),
//    shared by all 8 waves -> 8x less global B traffic, ds_load_b128 feeds.
//  - each wave: 16x64 tile = 4 f32 accumulators, one A fragment (two
//    global_load_b128) reused across 4 v_wmma_f32_16x16x32_bf16 per K-step.
// ---------------------------------------------------------------------------
__global__ __launch_bounds__(256) void gemm_bf16_wmma(
    const __bf16* __restrict__ Ab, const __bf16* __restrict__ Wt,
    const float* __restrict__ bias, float* __restrict__ C,
    int M, int N, int Kpad) {
  __shared__ __bf16 Bs[64 * (MAX_KPAD + LDS_PAD)];

  const int lane   = threadIdx.x & 31;
  const int wave   = threadIdx.x >> 5;
  const int tilesN = (N + 63) >> 6;
  const int bm     = blockIdx.x / tilesN;          // 128-row block index
  const int tn0    = (blockIdx.x % tilesN) << 6;   // 64-col panel base
  const int tm     = bm * 128 + wave * 16;         // wave-uniform
  const int lstr   = Kpad + LDS_PAD;

  // ---- cooperative stage of the B panel into LDS (16B vector copies) ------
  const int kc     = Kpad >> 3;                    // 8-bf16 chunks per row
  const int chunks = 64 * kc;
  for (int c = threadIdx.x; c < chunks; c += 256) {
    const int row = c / kc;
    const int off = (c - row * kc) << 3;
    *(v8bf*)(&Bs[row * lstr + off]) = *(const v8bf*)(Wt + (long)(tn0 + row) * Kpad + off);
  }
  __syncthreads();

  const int lh    = lane & 15;
  const int khalf = (lane >> 4) << 3;  // lanes 0-15: K 0-7/16-23; 16-31: 8-15/24-31
  const __bf16* __restrict__ arow = Ab + (long)(tm + lh) * Kpad + khalf;

  v8f acc[4] = {{}, {}, {}, {}};
  for (int kk = 0; kk < Kpad; kk += 32) {
    __builtin_prefetch(arow + kk + 64, 0, 0);  // next A chunks (speculative)
    union { v16bf v; v8bf h[2]; } a;
    a.h[0] = *(const v8bf*)(arow + kk);
    a.h[1] = *(const v8bf*)(arow + kk + 16);
#pragma unroll
    for (int t = 0; t < 4; ++t) {
      const __bf16* brow = &Bs[(t * 16 + lh) * lstr + khalf];
      union { v16bf v; v8bf h[2]; } b;
      b.h[0] = *(const v8bf*)(brow + kk);
      b.h[1] = *(const v8bf*)(brow + kk + 16);
      acc[t] = __builtin_amdgcn_wmma_f32_16x16x32_bf16(
          /*neg_a=*/false, a.v, /*neg_b=*/false, b.v,
          /*c_mod=*/(short)0, acc[t], /*reuse_a=*/false, /*reuse_b=*/false);
    }
  }

#pragma unroll
  for (int t = 0; t < 4; ++t) {
    const int nc = tn0 + t * 16 + lh;
    if (nc >= N) continue;
    const float bv = (bias != nullptr) ? bias[nc] : 0.0f;
#pragma unroll
    for (int r = 0; r < 8; ++r) {
      const int mr = tm + r + ((lane < 16) ? 0 : 8);  // C/D layout: VGPR r -> M=r / r+8
      if (mr < M) C[(long)mr * N + nc] = acc[t][r] + bv;
    }
  }
}

// ---------------------------------------------------------------------------
// Degree / normalization
// ---------------------------------------------------------------------------
__global__ void deg_init_k(float* deg, int N) {
  int i = blockIdx.x * blockDim.x + threadIdx.x;
  if (i < N) deg[i] = 1.0f;  // self-loop
}
__global__ void deg_edges_k(const int* __restrict__ dst, float* deg, int E) {
  int e = blockIdx.x * blockDim.x + threadIdx.x;
  if (e < E) atomicAdd(&deg[dst[e]], 1.0f);
}
__global__ void deg_finish_k(float* deg, int N) {
  int i = blockIdx.x * blockDim.x + threadIdx.x;
  if (i < N) {
    float d = deg[i];
    deg[i] = (d > 0.0f) ? rsqrtf(d) : 0.0f;
  }
}

// ---------------------------------------------------------------------------
// Aggregation: h[n] = t[n]*dinv[n]^2 + bias, then h[dst] += t[src]*norm
// (L2-resident atomics: 51 MB feature matrix fits the 192 MB L2)
// ---------------------------------------------------------------------------
__global__ void agg_init_k(const float* __restrict__ t, const float* __restrict__ dinv,
                           const float* __restrict__ bias, float* __restrict__ h, int F) {
  const int n = blockIdx.x;
  const int f = threadIdx.x;  // blockDim == F
  const float dv = dinv[n];
  h[(long)n * F + f] = t[(long)n * F + f] * dv * dv + bias[f];
}
__global__ void edge_scatter_k(const float* __restrict__ t, const int* __restrict__ src,
                               const int* __restrict__ dst, const float* __restrict__ dinv,
                               float* h, int F) {
  const int e = blockIdx.x;
  const int s = src[e];
  const int d = dst[e];
  const float nrm = dinv[s] * dinv[d];
  const int f = threadIdx.x;  // blockDim == F
  atomicAdd(&h[(long)d * F + f], t[(long)s * F + f] * nrm);
}
__global__ void relu_k(float* p, long n) {
  long i = (long)blockIdx.x * blockDim.x + threadIdx.x;
  if (i < n) p[i] = fmaxf(p[i], 0.0f);
}
__global__ void zero_k(float* p, int n) {
  int i = blockIdx.x * blockDim.x + threadIdx.x;
  if (i < n) p[i] = 0.0f;
}

// ---------------------------------------------------------------------------
// Global mean pool
// ---------------------------------------------------------------------------
__global__ void pool_accum_k(const float* __restrict__ h, const int* __restrict__ batch,
                             float* pooled, float* cnt, int F) {
  const int n = blockIdx.x;
  const int g = batch[n];
  const int f = threadIdx.x;  // blockDim == F
  atomicAdd(&pooled[(long)g * F + f], h[(long)n * F + f]);
  if (f == 0) atomicAdd(&cnt[g], 1.0f);
}
__global__ void pool_div_k(float* pooled, const float* cnt, int F) {
  const int g = blockIdx.x;
  const int f = threadIdx.x;
  pooled[(long)g * F + f] /= fmaxf(cnt[g], 1.0f);
}

// ---------------------------------------------------------------------------
// Row softmax: one wave32 per row
// ---------------------------------------------------------------------------
__global__ void softmax_rows_k(const float* __restrict__ in, float* __restrict__ out,
                               int rows, int cols) {
  const int wave = threadIdx.x >> 5;
  const int lane = threadIdx.x & 31;
  const int row  = blockIdx.x * (blockDim.x >> 5) + wave;
  if (row >= rows) return;
  float mx = -1e30f;
  for (int c = lane; c < cols; c += 32) mx = fmaxf(mx, in[(long)row * cols + c]);
#pragma unroll
  for (int o = 16; o > 0; o >>= 1) mx = fmaxf(mx, __shfl_xor(mx, o, 32));
  float sum = 0.0f;
  for (int c = lane; c < cols; c += 32) sum += __expf(in[(long)row * cols + c] - mx);
#pragma unroll
  for (int o = 16; o > 0; o >>= 1) sum += __shfl_xor(sum, o, 32);
  const float inv = 1.0f / sum;
  for (int c = lane; c < cols; c += 32)
    out[(long)row * cols + c] = __expf(in[(long)row * cols + c] - mx) * inv;
}

__global__ void concat_rows_k(const float* __restrict__ a, int ac,
                              const float* __restrict__ b, int bc,
                              float* __restrict__ out, int rows) {
  const int tot = ac + bc;
  int idx = blockIdx.x * blockDim.x + threadIdx.x;
  if (idx >= rows * tot) return;
  const int r = idx / tot;
  const int c = idx % tot;
  out[idx] = (c < ac) ? a[(long)r * ac + c] : b[(long)r * bc + (c - ac)];
}

// ---------------------------------------------------------------------------
// Orchestration: convert+pad operands, then WMMA GEMM
// ---------------------------------------------------------------------------
static inline void run_gemm(const float* A, const float* W, const float* bias, float* C,
                            int M, int K, int N, __bf16* Ab, __bf16* Wt, hipStream_t s) {
  const int Kpad = cdiv(K, 32) * 32;      // <= MAX_KPAD for this model
  const int Mpad = cdiv(M, 128) * 128;    // block covers 128 rows
  const int Npad = cdiv(N, 64) * 64;
  const long ta = (long)Mpad * Kpad;
  const long tw = (long)Npad * Kpad;
  cvt_a_bf16_k <<<(int)cdivl(ta, 256), 256, 0, s>>>(A, Ab, M, K, Kpad, ta);
  cvt_wt_bf16_k<<<(int)cdivl(tw, 256), 256, 0, s>>>(W, Wt, K, N, Kpad, tw);
  const int blocks = (Mpad / 128) * (Npad / 64);
  gemm_bf16_wmma<<<blocks, 256, 0, s>>>(Ab, Wt, bias, C, M, N, Kpad);
}

extern "C" void kernel_launch(void* const* d_in, const int* in_sizes, int n_in,
                              void* d_out, int out_size, void* d_ws, size_t ws_size,
                              hipStream_t stream) {
  const float* x     = (const float*)d_in[0];
  const int*   ei    = (const int*)d_in[1];
  const int*   batch = (const int*)d_in[2];
  const float* W1 = (const float*)d_in[3];  const float* b1 = (const float*)d_in[4];
  const float* W2 = (const float*)d_in[5];  const float* b2 = (const float*)d_in[6];
  const float* W3 = (const float*)d_in[7];  const float* b3 = (const float*)d_in[8];
  const float* Wc1 = (const float*)d_in[9];  const float* bc1 = (const float*)d_in[10];
  const float* Wc2 = (const float*)d_in[11]; const float* bc2 = (const float*)d_in[12];
  const float* Wc3 = (const float*)d_in[13]; const float* bc3 = (const float*)d_in[14];

  const int N   = in_sizes[2];                // 50000 nodes
  const int E   = in_sizes[1] / 2;            // 800000 edges
  const int HID = in_sizes[4];                // 256
  const int IND = in_sizes[3] / HID;          // 300
  const int C1  = in_sizes[10];               // 16
  const int C2  = in_sizes[12];               // 64
  const int C3  = in_sizes[14];               // 256
  const int G   = out_size / (C1 + C2 + C3);  // 256 graphs

  const int* src = ei;
  const int* dst = ei + E;

  // workspace layout (float units; bf16 buffers at the end, 16B-aligned)
  float* ws = (float*)d_ws;
  const long szH = (long)N * HID;
  float* bufT   = ws;                           // transformed  [N,HID]
  float* bufH   = bufT + szH;                   // aggregated   [N,HID]
  float* dinv   = bufH + szH;                   // [N]
  float* pooled = dinv + N;                     // [G,HID]
  float* cnt    = pooled + (long)G * HID;       // [G]
  float* p1     = cnt + G;                      // [G,C1]
  float* p2     = p1 + (long)G * C1;            // [G,C2]
  float* cat2   = p2 + (long)G * C2;            // [G,HID+C1]
  float* cat3   = cat2 + (long)G * (HID + C1);  // [G,HID+C2]
  float* fend   = cat3 + (long)G * (HID + C2);

  const int KpadMax = cdiv(IND > HID + C2 ? IND : HID + C2, 32) * 32;  // 320
  __bf16* Ab = (__bf16*)fend;                           // [Mpad128, KpadMax] worst case
  __bf16* Wt = Ab + (long)cdiv(N, 128) * 128 * KpadMax; // [Npad64, KpadMax]

  float* l1 = (float*)d_out;
  float* l2 = l1 + (long)G * C1;
  float* l3 = l2 + (long)G * C2;

  // --- symmetric GCN normalization -----------------------------------------
  deg_init_k  <<<cdiv(N, 256), 256, 0, stream>>>(dinv, N);
  deg_edges_k <<<cdiv(E, 256), 256, 0, stream>>>(dst, dinv, E);
  deg_finish_k<<<cdiv(N, 256), 256, 0, stream>>>(dinv, N);

  // --- layer 1: relu(agg(x @ W1) + b1) -------------------------------------
  run_gemm(x, W1, nullptr, bufT, N, IND, HID, Ab, Wt, stream);
  agg_init_k    <<<N, HID, 0, stream>>>(bufT, dinv, b1, bufH, HID);
  edge_scatter_k<<<E, HID, 0, stream>>>(bufT, src, dst, dinv, bufH, HID);
  relu_k<<<(int)cdivl(szH, 256), 256, 0, stream>>>(bufH, szH);

  // --- layer 2 -------------------------------------------------------------
  run_gemm(bufH, W2, nullptr, bufT, N, HID, HID, Ab, Wt, stream);
  agg_init_k    <<<N, HID, 0, stream>>>(bufT, dinv, b2, bufH, HID);
  edge_scatter_k<<<E, HID, 0, stream>>>(bufT, src, dst, dinv, bufH, HID);
  relu_k<<<(int)cdivl(szH, 256), 256, 0, stream>>>(bufH, szH);

  // --- layer 3 (no relu) ---------------------------------------------------
  run_gemm(bufH, W3, nullptr, bufT, N, HID, HID, Ab, Wt, stream);
  agg_init_k    <<<N, HID, 0, stream>>>(bufT, dinv, b3, bufH, HID);
  edge_scatter_k<<<E, HID, 0, stream>>>(bufT, src, dst, dinv, bufH, HID);

  // --- global mean pool ----------------------------------------------------
  zero_k<<<cdiv(G * HID, 256), 256, 0, stream>>>(pooled, G * HID);
  zero_k<<<cdiv(G, 256), 256, 0, stream>>>(cnt, G);
  pool_accum_k<<<N, HID, 0, stream>>>(bufH, batch, pooled, cnt, HID);
  pool_div_k  <<<G, HID, 0, stream>>>(pooled, cnt, HID);

  // --- hierarchical classifier ---------------------------------------------
  run_gemm(pooled, Wc1, bc1, l1, G, HID, C1, Ab, Wt, stream);
  softmax_rows_k<<<cdiv(G, 8), 256, 0, stream>>>(l1, p1, G, C1);

  concat_rows_k<<<cdiv(G * (HID + C1), 256), 256, 0, stream>>>(pooled, HID, p1, C1, cat2, G);
  run_gemm(cat2, Wc2, bc2, l2, G, HID + C1, C2, Ab, Wt, stream);
  softmax_rows_k<<<cdiv(G, 8), 256, 0, stream>>>(l2, p2, G, C2);

  concat_rows_k<<<cdiv(G * (HID + C2), 256), 256, 0, stream>>>(pooled, HID, p2, C2, cat3, G);
  run_gemm(cat3, Wc3, bc3, l3, G, HID + C2, C3, Ab, Wt, stream);
}
